// Head_5849745457514
// MI455X (gfx1250) — compile-verified, matrix operand order checked by reading, not run on hardware
//
#include <hip/hip_runtime.h>
#include <hip/hip_bf16.h>

// ---------------------------------------------------------------------------
// Causal attention head for MI455X (gfx1250, wave32, WMMA f32_16x16x32_f16)
//   B=8, T=4096, C=1024, H=64
// Kernel 0: convert/transpose weights -> fp16 Wt[mat][n][k]  (L2-resident)
// Kernel 1: QKV projection with double-buffered async W staging
// Kernel 2: flash attention, double-buffered async K/V tiles (ASYNCcnt DMA)
// ---------------------------------------------------------------------------

typedef _Float16 half8  __attribute__((ext_vector_type(8)));
typedef _Float16 v16h   __attribute__((ext_vector_type(16)));
typedef float    v8f    __attribute__((ext_vector_type(8)));
typedef float    f32x4  __attribute__((ext_vector_type(4)));

__device__ __forceinline__ v16h make_v16h(half8 lo, half8 hi) {
  v16h r;
#pragma unroll
  for (int i = 0; i < 8; ++i) { r[i] = lo[i]; r[i + 8] = hi[i]; }
  return r;
}

#define WMMA_F16(A, Bf, Cf) \
  __builtin_amdgcn_wmma_f32_16x16x32_f16(false, (A), false, (Bf), (short)0, (Cf), false, false)

// gfx1250 async global->LDS DMA (ASYNCcnt-tracked; no VGPR round trip).
// Inline asm is opaque to the compiler's counter bookkeeping, so waits are
// issued manually below.
__device__ __forceinline__ void async_b128(unsigned lds_addr, unsigned long long gaddr) {
  asm volatile("global_load_async_to_lds_b128 %0, %1, off"
               :: "v"(lds_addr), "v"(gaddr) : "memory");
}
#define WAIT_ASYNC(n) asm volatile("s_wait_asynccnt %0" :: "i"(n) : "memory")

__device__ __forceinline__ unsigned lds_addr_of(const void* p) {
  return (unsigned)(uintptr_t)p;  // low 32 bits of generic ptr = LDS offset
}

// ---------------------------------------------------------------------------
// Kernel 0: W[C,H] fp32 -> Wt[mat][n=0..63][k=0..1023] fp16 (k contiguous).
// One-shot 384KB transform; stays in L2 for kernel 1's async staging.
// ---------------------------------------------------------------------------
__global__ __launch_bounds__(256) void convert_w_kernel(
    const float* __restrict__ Wq, const float* __restrict__ Wk,
    const float* __restrict__ Wv, _Float16* __restrict__ Wt) {
  const int i   = blockIdx.x * 256 + threadIdx.x;  // 0 .. 3*65536-1
  const int mat = i >> 16;
  const int f   = i & 65535;                       // f = k*64 + n (coalesced read)
  const int k   = f >> 6;
  const int n   = f & 63;
  const float* W = (mat == 0) ? Wq : (mat == 1) ? Wk : Wv;
  Wt[((long)mat << 16) + n * 1024 + k] = (_Float16)W[f];
}

// ---------------------------------------------------------------------------
// Kernel 1: fused QKV projection.
// 128 threads (4 waves); each wave computes 16 rows x 64 cols of Q, K, V.
// Weight chunks [3][64n][32k] fp16 stream into LDS via double-buffered async
// DMA (6 x b128 per thread per chunk); A-fragments come straight from global
// x (float4 -> fp16).
// ---------------------------------------------------------------------------
__global__ __launch_bounds__(128) void proj_qkv_kernel(
    const float* __restrict__ x, const _Float16* __restrict__ Wtg,
    _Float16* __restrict__ Qh, _Float16* __restrict__ Kh,
    _Float16* __restrict__ Vt, int T) {
  constexpr int C = 1024;
  constexpr int H = 64;

  __shared__ alignas(16) _Float16 Wt[2][3][H * 32];  // [buf][mat][n*32+k]

  const int tid  = threadIdx.x;
  const int wave = tid >> 5;
  const int lane = tid & 31;
  const int l16  = lane & 15;
  const int off  = (lane >> 4) << 3;   // K sub-offset for A/B frags
  const int roff = (lane >> 4) << 3;   // row offset for C/D layout

  const long rowbase = (long)blockIdx.x * 64 + wave * 16;
  const long arow    = rowbase + l16;

  auto stageW = [&](int bb, int kb) {
#pragma unroll
    for (int it = 0; it < 6; ++it) {
      const int idx = tid + it * 128;        // 0..767
      const int mat = idx >> 8;              // 0..2
      const int rem = idx & 255;
      const int n   = rem >> 2;              // 0..63
      const int q   = rem & 3;               // 8-half group
      async_b128(lds_addr_of(&Wt[bb][mat][n * 32 + q * 8]),
                 (unsigned long long)(Wtg + ((long)mat << 16) + n * 1024 + kb + q * 8));
    }
  };

  v8f acc[3][4];
#pragma unroll
  for (int m = 0; m < 3; ++m)
#pragma unroll
    for (int n = 0; n < 4; ++n) acc[m][n] = {};

  stageW(0, 0);
  for (int kb = 0, c = 0; kb < C; kb += 32, ++c) {
    const int bb = c & 1;
    if (kb + 32 < C) { stageW(bb ^ 1, kb + 32); WAIT_ASYNC(6); }
    else             { WAIT_ASYNC(0); }
    __syncthreads();  // all waves' chunk-kb DMAs landed

    // ---- A fragment: x[arow, kb..kb+31] as fp16 --------------------------
    const float* xr = x + arow * C + kb;
    f32x4 a0 = *(const f32x4*)(xr + off);
    f32x4 a1 = *(const f32x4*)(xr + off + 4);
    f32x4 a2 = *(const f32x4*)(xr + off + 16);
    f32x4 a3 = *(const f32x4*)(xr + off + 20);
    v16h af;
#pragma unroll
    for (int i = 0; i < 4; ++i) {
      af[i]      = (_Float16)a0[i];
      af[i + 4]  = (_Float16)a1[i];
      af[i + 8]  = (_Float16)a2[i];
      af[i + 12] = (_Float16)a3[i];
    }

    // ---- 12 WMMAs per chunk ---------------------------------------------
#pragma unroll
    for (int mat = 0; mat < 3; ++mat) {
      const _Float16* wb = &Wt[bb][mat][0];
#pragma unroll
      for (int nt = 0; nt < 4; ++nt) {
        const int n = nt * 16 + l16;
        half8 b0 = *(const half8*)(wb + n * 32 + off);
        half8 b1 = *(const half8*)(wb + n * 32 + 16 + off);
        acc[mat][nt] = WMMA_F16(af, make_v16h(b0, b1), acc[mat][nt]);
      }
    }
    __syncthreads();  // done reading buf bb before it is refilled
  }

  // ---- store Q,K row-major fp16; V transposed [b][h][t] ------------------
#pragma unroll
  for (int nt = 0; nt < 4; ++nt) {
    const int col = nt * 16 + l16;
#pragma unroll
    for (int r = 0; r < 8; ++r) {
      const long grow = rowbase + r + roff;
      Qh[grow * H + col] = (_Float16)acc[0][nt][r];
      Kh[grow * H + col] = (_Float16)acc[1][nt][r];
      const long b = grow / T;
      const long t = grow - b * T;
      Vt[b * (long)H * T + (long)col * T + t] = (_Float16)acc[2][nt][r];
    }
  }
}

// ---------------------------------------------------------------------------
// Kernel 2: flash attention (causal, online softmax), double-buffered
// async K/V tile staging. 128 threads (4 waves) per (batch, 64-query block);
// each wave owns 16 query rows.
// ---------------------------------------------------------------------------
__global__ __launch_bounds__(128) void flash_attn_kernel(
    const _Float16* __restrict__ Qh, const _Float16* __restrict__ Kh,
    const _Float16* __restrict__ Vt, float* __restrict__ out, int T) {
  constexpr int H = 64;
  constexpr float SCALE = 0.125f;  // 1/sqrt(64)

  __shared__ alignas(16) _Float16 Klds[2][64 * H];      // [buf][key][h]
  __shared__ alignas(16) _Float16 Vlds[2][H * 64];      // [buf][h][key]
  __shared__ alignas(16) _Float16 Plds[4][16 * 64];     // per-wave P staging

  const int tid  = threadIdx.x;
  const int wave = tid >> 5;
  const int lane = tid & 31;
  const int l16  = lane & 15;
  const int off  = (lane >> 4) << 3;
  const int roff = (lane >> 4) << 3;

  const int qb = blockIdx.x;
  const int b  = blockIdx.y;
  const long base = (long)b * T;
  const int qrow0 = qb * 64 + wave * 16;

  auto stageKV = [&](int bb, int j) {
    const _Float16* kp = Kh + (base + (long)j * 64) * H;
    const _Float16* vp = Vt + (long)b * H * T + (long)j * 64;
#pragma unroll
    for (int it = 0; it < 4; ++it) {
      const int v = tid + it * 128;          // 0..511
      async_b128(lds_addr_of(&Klds[bb][v * 8]),
                 (unsigned long long)(kp + v * 8));
      const int h  = v >> 3;
      const int ko = (v & 7) * 8;
      async_b128(lds_addr_of(&Vlds[bb][h * 64 + ko]),
                 (unsigned long long)(vp + (long)h * T + ko));
    }
  };

  // ---- Q fragments (held for the whole kernel) ---------------------------
  v16h qf[2];
  {
    const _Float16* qp = Qh + (base + qrow0 + l16) * H;
#pragma unroll
    for (int kc = 0; kc < 2; ++kc) {
      half8 lo = *(const half8*)(qp + kc * 32 + off);
      half8 hi = *(const half8*)(qp + kc * 32 + 16 + off);
      qf[kc] = make_v16h(lo, hi);
    }
  }

  v8f o[4];
#pragma unroll
  for (int nt = 0; nt < 4; ++nt) o[nt] = {};
  float m[8], l[8];
#pragma unroll
  for (int r = 0; r < 8; ++r) { m[r] = -1.0e30f; l[r] = 0.0f; }

  stageKV(0, 0);
  for (int j = 0; j <= qb; ++j) {
    const int bb = j & 1;
    if (j < qb) { stageKV(bb ^ 1, j + 1); WAIT_ASYNC(8); }  // j's 8 DMAs done
    else        { WAIT_ASYNC(0); }
    __syncthreads();  // all waves' tile-j DMAs landed

    // ---- S = Q K^T -------------------------------------------------------
    v8f s[4];
#pragma unroll
    for (int nt = 0; nt < 4; ++nt) {
      s[nt] = {};
      const int key = nt * 16 + l16;
#pragma unroll
      for (int kc = 0; kc < 2; ++kc) {
        half8 b0 = *(const half8*)(&Klds[bb][key * 64 + kc * 32 + off]);
        half8 b1 = *(const half8*)(&Klds[bb][key * 64 + kc * 32 + 16 + off]);
        s[nt] = WMMA_F16(qf[kc], make_v16h(b0, b1), s[nt]);
      }
    }

    // ---- scale + causal mask on the diagonal block -----------------------
    const bool diag = (j == qb);
#pragma unroll
    for (int nt = 0; nt < 4; ++nt) {
      const int key = j * 64 + nt * 16 + l16;
#pragma unroll
      for (int r = 0; r < 8; ++r) {
        float v = s[nt][r] * SCALE;
        if (diag && key > (qrow0 + r + roff)) v = -1.0e30f;
        s[nt][r] = v;
      }
    }

    // ---- online softmax --------------------------------------------------
    float alpha[8];
#pragma unroll
    for (int r = 0; r < 8; ++r) {
      float v = fmaxf(fmaxf(s[0][r], s[1][r]), fmaxf(s[2][r], s[3][r]));
#pragma unroll
      for (int d = 1; d < 16; d <<= 1) v = fmaxf(v, __shfl_xor(v, d, 32));
      const float mnew = fmaxf(m[r], v);
      alpha[r] = __expf(m[r] - mnew);
      m[r] = mnew;
    }
#pragma unroll
    for (int r = 0; r < 8; ++r) {
      float rs = 0.0f;
#pragma unroll
      for (int nt = 0; nt < 4; ++nt) {
        const float p = __expf(s[nt][r] - m[r]);
        s[nt][r] = p;
        rs += p;
      }
#pragma unroll
      for (int d = 1; d < 16; d <<= 1) rs += __shfl_xor(rs, d, 32);
      l[r] = l[r] * alpha[r] + rs;
    }
#pragma unroll
    for (int nt = 0; nt < 4; ++nt)
#pragma unroll
      for (int r = 0; r < 8; ++r) o[nt][r] *= alpha[r];

    // ---- transpose P through per-wave LDS region (C/D -> A layout) -------
    _Float16* pw = &Plds[wave][0];
#pragma unroll
    for (int nt = 0; nt < 4; ++nt)
#pragma unroll
      for (int r = 0; r < 8; ++r)
        pw[(r + roff) * 64 + nt * 16 + l16] = (_Float16)s[nt][r];
    // LDS ops within a wave are in-order (DScnt) -> reads below see writes.

    v16h pf[2];
#pragma unroll
    for (int kc = 0; kc < 2; ++kc) {
      half8 lo = *(const half8*)(pw + l16 * 64 + kc * 32 + off);
      half8 hi = *(const half8*)(pw + l16 * 64 + kc * 32 + 16 + off);
      pf[kc] = make_v16h(lo, hi);
    }

    // ---- O += P V --------------------------------------------------------
#pragma unroll
    for (int nt = 0; nt < 4; ++nt) {
      const int h = nt * 16 + l16;
#pragma unroll
      for (int kc = 0; kc < 2; ++kc) {
        half8 b0 = *(const half8*)(&Vlds[bb][h * 64 + kc * 32 + off]);
        half8 b1 = *(const half8*)(&Vlds[bb][h * 64 + kc * 32 + 16 + off]);
        o[nt] = WMMA_F16(pf[kc], make_v16h(b0, b1), o[nt]);
      }
    }
    __syncthreads();  // all waves done reading buf bb before it refills
  }

  // ---- normalize and store fp32 output -----------------------------------
  float inv[8];
#pragma unroll
  for (int r = 0; r < 8; ++r) inv[r] = 1.0f / l[r];
#pragma unroll
  for (int nt = 0; nt < 4; ++nt) {
    const int h = nt * 16 + l16;
#pragma unroll
    for (int r = 0; r < 8; ++r) {
      const long q = base + qrow0 + r + roff;
      out[q * H + h] = o[nt][r] * inv[r];
    }
  }
}

// ---------------------------------------------------------------------------
extern "C" void kernel_launch(void* const* d_in, const int* in_sizes, int n_in,
                              void* d_out, int out_size, void* d_ws, size_t ws_size,
                              hipStream_t stream) {
  (void)in_sizes; (void)n_in; (void)out_size; (void)ws_size;
  const float* x  = (const float*)d_in[0];
  const float* Wq = (const float*)d_in[1];
  const float* Wk = (const float*)d_in[2];
  const float* Wv = (const float*)d_in[3];
  float* out = (float*)d_out;

  constexpr int Bn = 8, T = 4096, H = 64;
  const long N = (long)Bn * T;

  _Float16* Qh  = (_Float16*)d_ws;     // [N, H]
  _Float16* Kh  = Qh + N * H;          // [N, H]
  _Float16* Vt  = Kh + N * H;          // [Bn, H, T] (transposed V)
  _Float16* Wtg = Vt + N * H;          // [3, 64, 1024] fp16 transposed weights

  convert_w_kernel<<<dim3(3 * 65536 / 256), 256, 0, stream>>>(Wq, Wk, Wv, Wtg);
  proj_qkv_kernel<<<dim3((unsigned)(N / 64)), 128, 0, stream>>>(
      x, Wtg, Qh, Kh, Vt, T);
  flash_attn_kernel<<<dim3(T / 64, Bn), 128, 0, stream>>>(Qh, Kh, Vt, out, T);
}
